// MyConv2dRelu_21509196218421
// MI455X (gfx1250) — matrix operand order repaired and checked by reading, not compiled, not run on hardware
//
#include <hip/hip_runtime.h>
#include <hip/hip_bf16.h>

// ---------------------------------------------------------------------------
// Conv3x3 (SAME, NCHW, B=32, C=32, O=32, 224x224) with early-exit mask on the
// partial sum of the first 16 input channels, implemented as implicit GEMM on
// v_wmma_f32_16x16x32_bf16 (CDNA5 / gfx1250, wave32).
//
//   out = (partial<0) ? 0 : relu(partial + rest)
//
// K ordering per WMMA step (K=32): K = tapInPair*16 + channelIdx
//   steps 0..4  : channels  0..15, tap pairs (0,1)(2,3)(4,5)(6,7)(8,pad)
//   steps 5..9  : channels 16..31, same tap pairs
// B fragments are pre-swizzled into per-lane WMMA register order in d_ws and
// hoisted out of the row loop (loaded once per K-step, reused by 4 rows).
// For the pad half-step the A upper half reads tap 7 (valid, initialized
// data) and the corresponding B half is exactly zero, so the product is 0;
// this avoids p1==p0 register copies that trigger WMMA hazard NOPs.
// ---------------------------------------------------------------------------

typedef __attribute__((ext_vector_type(16))) __bf16 v16bf;
typedef __attribute__((ext_vector_type(8)))  __bf16 v8bf;
typedef __attribute__((ext_vector_type(8)))  float  v8f;

#define HW      224
#define CIN     32
#define COUT    32
#define TROWS   4            // output rows per workgroup
#define LROWS   (TROWS + 2)  // input rows staged (halo)
#define LCOLS   114          // 112 output cols + halo
#define CPITCH  40           // bf16 per (row,col) in LDS: 32 used + 8 pad (80B)

// ---- prep: swizzle W (f32 OIHW 3x3) into bf16 WMMA B-fragments ------------
// Layout: frag[step(10)][ntile(2)][lane(32)][16 bf16]  == flat index = tid
__global__ void prep_bfrag(const float* __restrict__ W,
                           unsigned short* __restrict__ Bfrag_u) {
    int tid = blockIdx.x * 256 + threadIdx.x;
    if (tid >= 10 * 2 * 32 * 16) return;
    int idx = tid & 15;          // element within lane (ascending K)
    int L   = (tid >> 4) & 31;   // lane
    int nt  = (tid >> 9) & 1;    // N tile
    int s   = tid >> 10;         // step 0..9
    int k     = (idx < 8) ? 8 * (L >> 4) + idx : 16 + 8 * (L >> 4) + (idx - 8);   // K row of B (table order: k = 8h+s / 16+8h+(s-8))
    int chalf = s / 5;
    int pair  = s % 5;
    int tap   = (k < 16) ? (2 * pair) : (2 * pair + 1);  // 9 == pad
    int c     = chalf * 16 + (k & 15);
    int o     = nt * 16 + (L & 15);          // N = lane%16
    float v = (tap <= 8) ? W[(o * CIN + c) * 9 + tap] : 0.0f;
    ((__bf16*)Bfrag_u)[tid] = (__bf16)v;
}

// ---- main conv kernel ------------------------------------------------------
__global__ __launch_bounds__(224)
void conv3x3_wmma(const float* __restrict__ x,
                  const unsigned short* __restrict__ Bfrag_u,
                  float* __restrict__ out) {
    __shared__ __align__(16) __bf16 tile[LROWS * LCOLS * CPITCH];

    const __bf16* Bfrag = (const __bf16*)Bfrag_u;
    const int x0 = blockIdx.x * 112;   // 0 or 112
    const int y0 = blockIdx.y * TROWS; // 0..220
    const int n  = blockIdx.z;         // batch
    const int tid  = threadIdx.x;
    const int lane = tid & 31;
    const int wave = tid >> 5;          // 0..6

    // ---- stage x tile into LDS as bf16, layout [row][col][channel] --------
    // (row,channel) pair per wave-iteration; lanes sweep columns (coalesced).
    for (int p = wave; p < LROWS * CIN; p += 7) {
        const int row = p >> 5;        // /CIN
        const int c   = p & (CIN - 1);
        const int gy  = y0 - 1 + row;
        const bool rowok = (unsigned)gy < (unsigned)HW;
        const float* xrow = x + ((n * CIN + c) * HW + (rowok ? gy : 0)) * HW;
        const int lbase = row * LCOLS * CPITCH + c;
#pragma unroll
        for (int cb = 0; cb < 4; ++cb) {
            const int col = lane + cb * 32;
            if (col < LCOLS) {
                const int gx = x0 - 1 + col;
                float v = (rowok && (unsigned)gx < (unsigned)HW) ? xrow[gx] : 0.0f;
                tile[lbase + col * CPITCH] = (__bf16)v;
            }
        }
    }
    __syncthreads();

    // ---- per-wave 16-pixel strip over TROWS rows --------------------------
    const int m  = lane & 15;         // A-matrix M row (pixel within strip)
    const int hi = (lane >> 4) & 1;   // lane group (K sub-block select)
    const int xw = wave * 16;         // strip base col (local)

    v8f acc[TROWS][4];                // [row][0,1]=partial N0/N1, [2,3]=rest
#pragma unroll
    for (int r = 0; r < TROWS; ++r)
#pragma unroll
        for (int q = 0; q < 4; ++q) acc[r][q] = v8f{};

#pragma unroll
    for (int s = 0; s < 10; ++s) {
        const int pair = s % 5;
        const int t0   = 2 * pair;
        // Pad half-step: read tap 7 (valid data); B is zero for those K rows.
        const int t1c  = (t0 + 1 > 8) ? 7 : (t0 + 1);
        const int coff = ((s < 5) ? 0 : 16) + hi * 8;
        const int ai   = (s < 5) ? 0 : 2;

        // B fragments for this K-step (invariant across rows)
        v16bf b0 = *(const v16bf*)(Bfrag + ((s * 2 + 0) * 32 + lane) * 16);
        v16bf b1 = *(const v16bf*)(Bfrag + ((s * 2 + 1) * 32 + lane) * 16);

#pragma unroll
        for (int r = 0; r < TROWS; ++r) {
            // A fragment: two 16B LDS loads (8 consecutive channels @ one tap)
            const int a0off = ((r + t0 / 3) * LCOLS + (xw + m + t0 % 3)) * CPITCH + coff;
            const int a1off = ((r + t1c / 3) * LCOLS + (xw + m + t1c % 3)) * CPITCH + coff;
            v8bf p0 = *(const v8bf*)(tile + a0off);
            v8bf p1 = *(const v8bf*)(tile + a1off);
            v16bf a = __builtin_shufflevector(p0, p1,
                0, 1, 2, 3, 4, 5, 6, 7, 8, 9, 10, 11, 12, 13, 14, 15);

            acc[r][ai + 0] = __builtin_amdgcn_wmma_f32_16x16x32_bf16(
                false, a, false, b0, (short)0, acc[r][ai + 0], false, false);
            acc[r][ai + 1] = __builtin_amdgcn_wmma_f32_16x16x32_bf16(
                false, a, false, b1, (short)0, acc[r][ai + 1], false, false);
        }
    }

    // ---- epilogue: mask on partial, relu on full, store -------------------
#pragma unroll
    for (int r = 0; r < TROWS; ++r) {
#pragma unroll
        for (int nt = 0; nt < 2; ++nt) {
            v8f part = acc[r][nt];
            v8f full = acc[r][nt] + acc[r][nt + 2];
            v8f res;
#pragma unroll
            for (int i = 0; i < 8; ++i) {
                float f = full[i] > 0.0f ? full[i] : 0.0f;
                res[i] = (part[i] < 0.0f) ? 0.0f : f;
            }
            // D layout: lane = N (channel), VGPR j = pixel j + hi*8
            const int o  = nt * 16 + m;
            const int gx = x0 + xw + hi * 8;
            float* p = out + ((n * COUT + o) * HW + (y0 + r)) * HW + gx;
            *(v8f*)p = res;   // 32B contiguous per lane
        }
    }
}

extern "C" void kernel_launch(void* const* d_in, const int* in_sizes, int n_in,
                              void* d_out, int out_size, void* d_ws, size_t ws_size,
                              hipStream_t stream) {
    const float* x = (const float*)d_in[0];   // [32,32,224,224] f32
    const float* W = (const float*)d_in[1];   // [32,32,3,3] f32
    float* out = (float*)d_out;               // [32,32,224,224] f32
    unsigned short* Bfrag = (unsigned short*)d_ws; // 10*2*32*16 bf16 = 20 KB

    prep_bfrag<<<40, 256, 0, stream>>>(W, Bfrag);

    dim3 grid(2, HW / TROWS, 32);             // (x halves, row blocks, batch)
    conv3x3_wmma<<<grid, 224, 0, stream>>>(x, Bfrag, out);
}
